// KRNN3_58428735095364
// MI455X (gfx1250) — matrix operand order repaired and checked by reading, not compiled, not run on hardware
//
#include <hip/hip_runtime.h>
#include <hip/hip_bf16.h>

typedef __attribute__((ext_vector_type(16))) __bf16 v16bf;
typedef __attribute__((ext_vector_type(8)))  float  v8f;

#define BN_TOTAL 131072
#define ROWS 64
#define TDIM 12
#define HDIM 64
#define G3   192
#define CCH  10
#define TOUT 12

// dynamic smem carve (bytes)
#define OFF_OUT   0                 // float [ROWS][12][10]  30720
#define OFF_XT    30720             // float [ROWS][12][2]    6144
#define OFF_HV    36864             // float [12][ROWS][2]    6144
#define OFF_WBE   43008             // bf16  [2][12][32][16] 24576
#define OFF_WBD   67584             // bf16  [2][12][32][16] 24576
#define OFF_HB    92160             // bf16  [4 waves][16][64] 8192
#define SMEM_BYTES 100352

// CDNA5 hardware transcendental tanh (TRANS unit, co-executes with XDL WMMA).
// ISA 7.4: one independent op / V_NOP required between a TRANS op and use of
// its result; the compiler can't see through inline asm, so embed v_nop.
__device__ __forceinline__ float tanh_hw(float x) {
  float r;
  asm("v_tanh_f32 %0, %1\n\tv_nop" : "=v"(r) : "v"(x));
  return r;
}
// sigmoid(x) = 0.5*tanh(0.5*x) + 0.5  -> 1 trans + mul + fma
__device__ __forceinline__ float sigmoid_hw(float x) {
  return fmaf(0.5f, tanh_hw(0.5f * x), 0.5f);
}

// A fragment (16x32 bf16): lane L holds row m=L%16, two 8-element K-runs
// run0 = kt*32 + hi*8 .. +7 ; run1 = run0 + 16  (per ISA 16-bit A layout)
__device__ __forceinline__ v16bf load_a_frag(const __bf16* hb, int lc, int hi, int kt) {
  const __bf16* p0 = hb + lc * HDIM + kt * 32 + hi * 8;
  const __bf16* p1 = p0 + 16;
  v16bf r;
#pragma unroll
  for (int i = 0; i < 8; ++i) { r[i] = p0[i]; r[8 + i] = p1[i]; }
  return r;
}

// B fragment: pre-swizzled in LDS so each lane reads 16 contiguous bf16
__device__ __forceinline__ v16bf load_b_frag(const __bf16* wb, int kt, int nt, int lane) {
  const __bf16* p = wb + (((kt * 12 + nt) * 32) + lane) * 16;
  v16bf r;
#pragma unroll
  for (int i = 0; i < 16; ++i) r[i] = p[i];
  return r;
}

__global__ __launch_bounds__(128) void krnn_fused(
    const float* __restrict__ X,
    const float* __restrict__ enc_w_ih, const float* __restrict__ enc_w_hh,
    const float* __restrict__ enc_b_ih, const float* __restrict__ enc_b_hh,
    const float* __restrict__ dec_w_ih, const float* __restrict__ dec_w_hh,
    const float* __restrict__ dec_b_ih, const float* __restrict__ dec_b_hh,
    const float* __restrict__ lin1_w, const float* __restrict__ lin1_b,
    const float* __restrict__ lin2_w, const float* __restrict__ lin2_b,
    const float* __restrict__ embed1,
    float* __restrict__ out)
{
  extern __shared__ char smem[];
  float*  out_lds = (float*)(smem + OFF_OUT);
  float*  xtile   = (float*)(smem + OFF_XT);
  float*  hvbuf   = (float*)(smem + OFF_HV);
  __bf16* wBe     = (__bf16*)(smem + OFF_WBE);
  __bf16* wBd     = (__bf16*)(smem + OFF_WBD);
  __bf16* hball   = (__bf16*)(smem + OFF_HB);

  const int tid  = threadIdx.x;
  const int wave = tid >> 5;
  const int lane = tid & 31;
  const int lc   = lane & 15;   // column within 16-wide tile / row for A frag
  const int hi   = lane >> 4;   // half-wave select
  const int bn0  = blockIdx.x * ROWS;

  __bf16* hb = hball + wave * 16 * HDIM;   // wave-private h staging (16 rows x 64)

  // stage X tile once (reused for hv, last0, query)
  for (int idx = tid; idx < ROWS * TDIM * 2; idx += 128)
    xtile[idx] = X[(size_t)bn0 * (TDIM * 2) + idx];
  __syncthreads();

  for (int ch = 0; ch < CCH; ++ch) {
    const int sz = ch + 3;   // SIZES = 3..12
    __syncthreads();         // protect previous channel's LDS reads

    // ---- swizzle w_hh (enc+dec) into WMMA B-fragment layout, fp32->bf16 ----
    {
      const float* whhE = enc_w_hh + ch * G3 * HDIM;
      const float* whhD = dec_w_hh + ch * G3 * HDIM;
      for (int idx = tid; idx < 2 * 12 * 32 * 16; idx += 128) {
        int e = idx & 15;
        int L = (idx >> 4) & 31;
        int rest = idx >> 9;
        int nt = rest % 12;
        int kt = rest / 12;
        int g = 16 * nt + (L & 15);
        int k = kt * 32 + 8 * (L >> 4) + (e & 7) + ((e >> 3) ? 16 : 0);   // B(k,n)=w_hh[g][k]
        wBe[idx] = (__bf16)whhE[g * HDIM + k];
        wBd[idx] = (__bf16)whhD[g * HDIM + k];
      }
    }
    // ---- hv[s][rl][f] = sum_t X[rl,t,f]*w2[s,t] + b2[s] ----
    {
      const float* w2 = lin2_w + ch * (TDIM * TDIM);
      const float* b2 = lin2_b + ch * TDIM;
      for (int idx = tid; idx < sz * ROWS * 2; idx += 128) {
        int f = idx & 1;
        int rl = (idx >> 1) & (ROWS - 1);
        int s = idx >> 7;
        float a = b2[s];
#pragma unroll
        for (int t = 0; t < TDIM; ++t)
          a += xtile[(rl * TDIM + t) * 2 + f] * w2[s * TDIM + t];
        hvbuf[idx] = a;
      }
    }
    // zero wave-private h staging
    for (int i = lane; i < 16 * HDIM; i += 32) hb[i] = (__bf16)0.0f;
    __syncthreads();

    // ---- hoist per-lane gate constants (encoder) ----
    const float* wihE = enc_w_ih + ch * G3 * 2;
    const float* bihE = enc_b_ih + ch * G3;
    const float* bhhE = enc_b_hh + ch * G3;
    float wr0[4], wr1[4], wz0[4], wz1[4], wn0[4], wn1[4];
    float bir[4], biz[4], bin_[4], bhr[4], bhz[4], bhn[4];
#pragma unroll
    for (int j = 0; j < 4; ++j) {
      int gr = 16 * j + lc;
      wr0[j] = wihE[gr * 2];           wr1[j] = wihE[gr * 2 + 1];
      wz0[j] = wihE[(gr + 64) * 2];    wz1[j] = wihE[(gr + 64) * 2 + 1];
      wn0[j] = wihE[(gr + 128) * 2];   wn1[j] = wihE[(gr + 128) * 2 + 1];
      bir[j] = bihE[gr]; biz[j] = bihE[gr + 64]; bin_[j] = bihE[gr + 128];
      bhr[j] = bhhE[gr]; bhz[j] = bhhE[gr + 64]; bhn[j] = bhhE[gr + 128];
    }

    float hreg[4][8];   // h tiles: element (m=p+8*hi, hidx=16*j+lc)
#pragma unroll
    for (int j = 0; j < 4; ++j)
#pragma unroll
      for (int p = 0; p < 8; ++p) hreg[j][p] = 0.0f;

    // ================= encoder: sz GRU steps =================
#pragma unroll 1
    for (int s = 0; s < sz; ++s) {
      v8f acc[12];
#pragma unroll
      for (int j = 0; j < 4; ++j)
#pragma unroll
        for (int p = 0; p < 8; ++p) {
          acc[j][p] = bhr[j]; acc[j + 4][p] = bhz[j]; acc[j + 8][p] = bhn[j];
        }
#pragma unroll
      for (int kt = 0; kt < 2; ++kt) {
        v16bf a = load_a_frag(hb, lc, hi, kt);
#pragma unroll
        for (int nt = 0; nt < 12; ++nt) {
          v16bf b = load_b_frag(wBe, kt, nt, lane);
          acc[nt] = __builtin_amdgcn_wmma_f32_16x16x32_bf16(
              false, a, false, b, (short)0, acc[nt], false, false);
        }
      }
      const float* hvs = hvbuf + s * (ROWS * 2);
#pragma unroll
      for (int p = 0; p < 8; ++p) {
        int rl = wave * 16 + p + 8 * hi;
        float x0 = hvs[rl * 2], x1 = hvs[rl * 2 + 1];
#pragma unroll
        for (int j = 0; j < 4; ++j) {
          float r = sigmoid_hw(x0 * wr0[j] + x1 * wr1[j] + bir[j] + acc[j][p]);
          float z = sigmoid_hw(x0 * wz0[j] + x1 * wz1[j] + biz[j] + acc[j + 4][p]);
          float n = tanh_hw(x0 * wn0[j] + x1 * wn1[j] + bin_[j] + r * acc[j + 8][p]);
          hreg[j][p] = (1.0f - z) * n + z * hreg[j][p];
        }
      }
#pragma unroll
      for (int j = 0; j < 4; ++j)
#pragma unroll
        for (int p = 0; p < 8; ++p)
          hb[(p + 8 * hi) * HDIM + 16 * j + lc] = (__bf16)hreg[j][p];
    }

    // ---- hoist decoder constants ----
    const float* wihD = dec_w_ih + ch * G3;
    const float* bihD = dec_b_ih + ch * G3;
    const float* bhhD = dec_b_hh + ch * G3;
    float widr[4], widz[4], widn[4], bdr[4], bdz[4], bdn[4], dhr[4], dhz[4], dhn[4], lw[4];
#pragma unroll
    for (int j = 0; j < 4; ++j) {
      int gr = 16 * j + lc;
      widr[j] = wihD[gr]; widz[j] = wihD[gr + 64]; widn[j] = wihD[gr + 128];
      bdr[j] = bihD[gr];  bdz[j] = bihD[gr + 64];  bdn[j] = bihD[gr + 128];
      dhr[j] = bhhD[gr];  dhz[j] = bhhD[gr + 64];  dhn[j] = bhhD[gr + 128];
      lw[j]  = lin1_w[ch * HDIM + gr];
    }
    const float l1b = lin1_b[ch];

    float lv[8];   // decoder scalar input, init = last0 = X[:, t=11, f=0]
#pragma unroll
    for (int p = 0; p < 8; ++p)
      lv[p] = xtile[((wave * 16 + p + 8 * hi) * TDIM + (TDIM - 1)) * 2];

    // ================= decoder: 12 GRU steps =================
#pragma unroll 1
    for (int t = 0; t < TOUT; ++t) {
      v8f acc[12];
#pragma unroll
      for (int j = 0; j < 4; ++j)
#pragma unroll
        for (int p = 0; p < 8; ++p) {
          acc[j][p] = dhr[j]; acc[j + 4][p] = dhz[j]; acc[j + 8][p] = dhn[j];
        }
#pragma unroll
      for (int kt = 0; kt < 2; ++kt) {
        v16bf a = load_a_frag(hb, lc, hi, kt);
#pragma unroll
        for (int nt = 0; nt < 12; ++nt) {
          v16bf b = load_b_frag(wBd, kt, nt, lane);
          acc[nt] = __builtin_amdgcn_wmma_f32_16x16x32_bf16(
              false, a, false, b, (short)0, acc[nt], false, false);
        }
      }
#pragma unroll
      for (int p = 0; p < 8; ++p) {
#pragma unroll
        for (int j = 0; j < 4; ++j) {
          float r = sigmoid_hw(lv[p] * widr[j] + bdr[j] + acc[j][p]);
          float z = sigmoid_hw(lv[p] * widz[j] + bdz[j] + acc[j + 4][p]);
          float n = tanh_hw(lv[p] * widn[j] + bdn[j] + r * acc[j + 8][p]);
          hreg[j][p] = (1.0f - z) * n + z * hreg[j][p];
        }
      }
#pragma unroll
      for (int j = 0; j < 4; ++j)
#pragma unroll
        for (int p = 0; p < 8; ++p)
          hb[(p + 8 * hi) * HDIM + 16 * j + lc] = (__bf16)hreg[j][p];

      // v = h . lin1_w + b : partial per lane, reduce across 16-lane half
      float part[8];
#pragma unroll
      for (int p = 0; p < 8; ++p) {
        float s4 = 0.0f;
#pragma unroll
        for (int j = 0; j < 4; ++j) s4 += hreg[j][p] * lw[j];
        part[p] = s4;
      }
#pragma unroll
      for (int off = 1; off < 16; off <<= 1)
#pragma unroll
        for (int p = 0; p < 8; ++p)
          part[p] += __shfl_xor(part[p], off, 32);
#pragma unroll
      for (int p = 0; p < 8; ++p) {
        float v = part[p] + l1b;
        lv[p] = v;
        if (lc == 0)
          out_lds[((wave * 16 + p + 8 * hi) * TOUT + t) * CCH + ch] = v;
      }
    }
  } // channels
  __syncthreads();

  // ===== fused attention combine: softmax over channels, weighted sum =====
  if (tid < ROWS) {
    const int rl = tid;
    const size_t bn = (size_t)bn0 + rl;
    float wq[TOUT];
#pragma unroll
    for (int o = 0; o < TOUT; ++o) {
      float s = 0.0f;
#pragma unroll
      for (int t = 0; t < TDIM; ++t)
        s += xtile[(rl * TDIM + t) * 2] * embed1[t * TOUT + o];
      wq[o] = s;
    }
    float wc[CCH], m = -1e30f;
#pragma unroll
    for (int c2 = 0; c2 < CCH; ++c2) {
      float s = 0.0f;
#pragma unroll
      for (int o = 0; o < TOUT; ++o)
        s += wq[o] * out_lds[(rl * TOUT + o) * CCH + c2];
      wc[c2] = s;
      m = fmaxf(m, s);
    }
    float se = 0.0f;
#pragma unroll
    for (int c2 = 0; c2 < CCH; ++c2) { wc[c2] = __expf(wc[c2] - m); se += wc[c2]; }
    float inv = 1.0f / se;
#pragma unroll
    for (int o = 0; o < TOUT; ++o) {
      float s = 0.0f;
#pragma unroll
      for (int c2 = 0; c2 < CCH; ++c2)
        s += out_lds[(rl * TOUT + o) * CCH + c2] * wc[c2];
      out[bn * TOUT + o] = s * inv;
    }
  }
}

extern "C" void kernel_launch(void* const* d_in, const int* in_sizes, int n_in,
                              void* d_out, int out_size, void* d_ws, size_t ws_size,
                              hipStream_t stream) {
  // d_in order: A(unused), X, enc_w_ih, enc_w_hh, enc_b_ih, enc_b_hh,
  //             dec_w_ih, dec_w_hh, dec_b_ih, dec_b_hh,
  //             lin1_w, lin1_b, lin2_w, lin2_b, embed1
  const float* X        = (const float*)d_in[1];
  const float* enc_w_ih = (const float*)d_in[2];
  const float* enc_w_hh = (const float*)d_in[3];
  const float* enc_b_ih = (const float*)d_in[4];
  const float* enc_b_hh = (const float*)d_in[5];
  const float* dec_w_ih = (const float*)d_in[6];
  const float* dec_w_hh = (const float*)d_in[7];
  const float* dec_b_ih = (const float*)d_in[8];
  const float* dec_b_hh = (const float*)d_in[9];
  const float* lin1_w   = (const float*)d_in[10];
  const float* lin1_b   = (const float*)d_in[11];
  const float* lin2_w   = (const float*)d_in[12];
  const float* lin2_b   = (const float*)d_in[13];
  const float* embed1   = (const float*)d_in[14];
  float* outp = (float*)d_out;

  dim3 grid(BN_TOTAL / ROWS);
  dim3 block(128);
  krnn_fused<<<grid, block, SMEM_BYTES, stream>>>(
      X, enc_w_ih, enc_w_hh, enc_b_ih, enc_b_hh,
      dec_w_ih, dec_w_hh, dec_b_ih, dec_b_hh,
      lin1_w, lin1_b, lin2_w, lin2_b, embed1, outp);
}
